// Block_2516850836150
// MI455X (gfx1250) — matrix operand order repaired and checked by reading, not compiled, not run on hardware
//
#include <hip/hip_runtime.h>

// ---------------------------------------------------------------------------
// Transformer block forward for MI455X (gfx1250), bf16 WMMA pipeline with
// CDNA5 async global->LDS copies (ASYNCcnt) and double-buffered LDS.
// B=8, T=1024, C=1024, H=16, D=64
// ---------------------------------------------------------------------------

typedef __attribute__((ext_vector_type(8)))  float  v8f;
typedef __attribute__((ext_vector_type(8)))  __bf16 v8bf;
typedef __attribute__((ext_vector_type(16))) __bf16 v16bf;

union Frag { v16bf v; v8bf h[2]; };

static constexpr int Bb = 8;
static constexpr int Tt = 1024;
static constexpr int Cc = 1024;
static constexpr int Hh = 16;
static constexpr int Dd = 64;
static constexpr int BT = Bb * Tt;   // 8192 rows

// Async copy of 16 bytes global -> LDS, tracked by ASYNCcnt (CDNA5).
static __device__ __forceinline__ void async_copy_b128(void* lds_ptr, const void* gptr) {
  unsigned lds = (unsigned)(uintptr_t)lds_ptr;   // low 32 bits = LDS offset
  asm volatile("global_load_async_to_lds_b128 %0, %1, off"
               :: "v"(lds), "v"(gptr) : "memory");
}
static __device__ __forceinline__ void wait_async() {
  asm volatile("s_wait_asynccnt 0x0" ::: "memory");
}

// ---------------------------------------------------------------------------
// fp32 -> bf16 conversion (weights)
// ---------------------------------------------------------------------------
__global__ void cvt_f32_bf16(const float* __restrict__ src,
                             __bf16* __restrict__ dst, int n) {
  int i = blockIdx.x * blockDim.x + threadIdx.x;
  if (i < n) dst[i] = (__bf16)src[i];
}

// ---------------------------------------------------------------------------
// LayerNorm: one block per row of 1024, output bf16
// ---------------------------------------------------------------------------
__global__ __launch_bounds__(256)
void ln_kernel(const float* __restrict__ x, const float* __restrict__ w,
               const float* __restrict__ b, __bf16* __restrict__ out) {
  __shared__ float red[256];
  const int row = blockIdx.x;
  const int tid = threadIdx.x;
  const float* xr = x + (size_t)row * Cc;

  float v[4];
  float s = 0.f;
#pragma unroll
  for (int i = 0; i < 4; ++i) { v[i] = xr[tid + i * 256]; s += v[i]; }
  red[tid] = s; __syncthreads();
  for (int k = 128; k > 0; k >>= 1) { if (tid < k) red[tid] += red[tid + k]; __syncthreads(); }
  float mean = red[0] * (1.f / Cc);
  __syncthreads();

  s = 0.f;
#pragma unroll
  for (int i = 0; i < 4; ++i) { float d = v[i] - mean; s += d * d; }
  red[tid] = s; __syncthreads();
  for (int k = 128; k > 0; k >>= 1) { if (tid < k) red[tid] += red[tid + k]; __syncthreads(); }
  float rstd = rsqrtf(red[0] * (1.f / Cc) + 1e-5f);

#pragma unroll
  for (int i = 0; i < 4; ++i) {
    int col = tid + i * 256;
    out[(size_t)row * Cc + col] = (__bf16)((v[i] - mean) * rstd * w[col] + b[col]);
  }
}

// ---------------------------------------------------------------------------
// bf16 GEMM with WMMA: C[M,N] = A[M,K] * B[K,N] (+bias, +GELU, +fp32 residual)
// Block tile 128x128, BK=32. 256 threads = 8 waves, wave tile 32x64 (2x4 WMMA).
// Double-buffered LDS; A tile copied with global_load_async_to_lds_b128,
// B tile loaded to regs (latency hidden behind WMMAs) and stored transposed.
// ---------------------------------------------------------------------------
template<bool BIAS, bool RES, bool GELU_, bool OUTBF>
__global__ __launch_bounds__(256)
void gemm_bf16_kernel(const __bf16* __restrict__ A, const __bf16* __restrict__ Bw,
                      const float* __restrict__ bias, const float* __restrict__ res,
                      float* __restrict__ outF, __bf16* __restrict__ outB,
                      int M, int N, int K) {
  __shared__ __align__(16) __bf16 As[2][128 * 40];
  __shared__ __align__(16) __bf16 Bt[2][128 * 40];

  const int tid  = threadIdx.x;
  const int lane = tid & 31;
  const int wave = tid >> 5;
  const int wm   = wave & 3;      // 4 waves along M
  const int wn   = wave >> 2;     // 2 waves along N
  const int row0 = blockIdx.y * 128;
  const int col0 = blockIdx.x * 128;

  const int lr  = lane & 15;
  const int hi  = lane >> 4;
  const int kbA = hi * 8;         // A-frag K split: {kb..kb+7, kb+16..kb+23}
  const int kbB = hi * 16;        // unused; B-frags load with kbA, same K split as A

  const v8f vzero = {0.f, 0.f, 0.f, 0.f, 0.f, 0.f, 0.f, 0.f};
  v8f acc[2][4];
#pragma unroll
  for (int i = 0; i < 2; ++i)
#pragma unroll
    for (int j = 0; j < 4; ++j) acc[i][j] = vzero;

  // per-thread copy coordinates
  const int ar0 = tid >> 2,          akc0 = (tid & 3) * 8;          // A chunk 0
  const int ar1 = (tid + 256) >> 2,  akc1 = ((tid + 256) & 3) * 8;  // A chunk 1
  const int bk0 = tid >> 4,          bcc0 = (tid & 15) * 8;         // B chunk 0
  const int bk1 = (tid + 256) >> 4,  bcc1 = ((tid + 256) & 15) * 8; // B chunk 1

  auto issue_async_A = [&](int buf, int k0) {
    async_copy_b128(&As[buf][ar0 * 40 + akc0], A + (size_t)(row0 + ar0) * K + k0 + akc0);
    async_copy_b128(&As[buf][ar1 * 40 + akc1], A + (size_t)(row0 + ar1) * K + k0 + akc1);
  };
  auto load_B = [&](int k0, v8bf* breg) {
    breg[0] = *(const v8bf*)(Bw + (size_t)(k0 + bk0) * N + col0 + bcc0);
    breg[1] = *(const v8bf*)(Bw + (size_t)(k0 + bk1) * N + col0 + bcc1);
  };
  auto store_B = [&](int buf, const v8bf* breg) {
#pragma unroll
    for (int j = 0; j < 8; ++j) Bt[buf][(bcc0 + j) * 40 + bk0] = breg[0][j];
#pragma unroll
    for (int j = 0; j < 8; ++j) Bt[buf][(bcc1 + j) * 40 + bk1] = breg[1][j];
  };

  // ---- prologue: fill buffer 0 ----
  v8bf breg[2];
  issue_async_A(0, 0);
  load_B(0, breg);
  store_B(0, breg);
  wait_async();
  __syncthreads();

  int buf = 0;
  for (int k0 = 0; k0 < K; k0 += 32) {
    const int k1 = k0 + 32;
    const bool more = (k1 < K);
    if (more) {               // kick off next tile before computing this one
      issue_async_A(buf ^ 1, k1);
      load_B(k1, breg);
    }

    Frag a[2], bfr[4];
#pragma unroll
    for (int mt = 0; mt < 2; ++mt) {
      const __bf16* p = As[buf] + (wm * 32 + mt * 16 + lr) * 40;
      a[mt].h[0] = *(const v8bf*)(p + kbA);
      a[mt].h[1] = *(const v8bf*)(p + kbA + 16);
    }
#pragma unroll
    for (int nt = 0; nt < 4; ++nt) {
      const __bf16* p = Bt[buf] + (wn * 64 + nt * 16 + lr) * 40;
      bfr[nt].h[0] = *(const v8bf*)(p + kbA);
      bfr[nt].h[1] = *(const v8bf*)(p + kbA + 16);
    }
#pragma unroll
    for (int mt = 0; mt < 2; ++mt)
#pragma unroll
      for (int nt = 0; nt < 4; ++nt)
        acc[mt][nt] = __builtin_amdgcn_wmma_f32_16x16x32_bf16(
            false, a[mt].v, false, bfr[nt].v, (short)0, acc[mt][nt], false, false);

    if (more) {
      store_B(buf ^ 1, breg);  // global-load wait lands here, after the WMMAs
      wait_async();
    }
    __syncthreads();
    buf ^= 1;
  }

  // Epilogue: C layout — lane holds col (lane&15), rows v + 8*(lane>>4)
#pragma unroll
  for (int mt = 0; mt < 2; ++mt) {
#pragma unroll
    for (int nt = 0; nt < 4; ++nt) {
      int col = col0 + wn * 64 + nt * 16 + lr;
      float bv = BIAS ? bias[col] : 0.f;
#pragma unroll
      for (int v = 0; v < 8; ++v) {
        int row = row0 + wm * 32 + mt * 16 + v + 8 * hi;
        float xv = acc[mt][nt][v] + bv;
        if (GELU_) {
          float t = xv;
          xv = 0.5f * t * (1.f + tanhf(0.7978845608f * (t + 0.044715f * t * t * t)));
        }
        if (RES) xv += res[(size_t)row * N + col];
        if (OUTBF) outB[(size_t)row * N + col] = (__bf16)xv;
        else       outF[(size_t)row * N + col] = xv;
      }
    }
  }
}

// ---------------------------------------------------------------------------
// Flash attention, causal, scale = 1/sqrt(C) = 1/32 (faithful to reference).
// Grid: B*H*(T/64) blocks of 128 threads (4 waves). Wave handles 16 q rows.
// Double-buffered K/V tiles: K copied async global->LDS (row-major, d is the
// contraction and contiguous for QK^T B-frags), V through regs, transposed
// into LDS for PV B-frags. P round-trips per-wave LDS to reach A layout.
// ---------------------------------------------------------------------------
__global__ __launch_bounds__(128)
void attn_kernel(const __bf16* __restrict__ qkv, __bf16* __restrict__ y) {
  __shared__ __align__(16) __bf16 Ks[2][32 * 72];      // [kpos][d]
  __shared__ __align__(16) __bf16 Vs[2][64 * 40];      // [d][kpos]
  __shared__ __align__(16) __bf16 Ps[4][16 * 40];      // per-wave P 16x32

  const int tid  = threadIdx.x;
  const int lane = tid & 31;
  const int wave = tid >> 5;
  const int nqt  = Tt / 64;
  const int qt   = blockIdx.x % nqt;
  const int bh   = blockIdx.x / nqt;
  const int h    = bh % Hh;
  const int bb   = bh / Hh;
  const int qbase = qt * 64 + wave * 16;
  const int lr  = lane & 15;
  const int hi  = lane >> 4;
  const int kbA = hi * 8;
  const int kbB = hi * 16;
  const float scale = 0.03125f;  // 1/sqrt(1024)

  // per-thread K/V copy coordinates (32x64 tile, 2 chunks of 8 bf16 each)
  const int kr0 = tid >> 3,          dc0 = (tid & 7) * 8;
  const int kr1 = (tid + 128) >> 3,  dc1 = ((tid + 128) & 7) * 8;

  auto issue_async_K = [&](int buf, int k0) {
    const __bf16* base = qkv + (size_t)(bb * Tt + k0) * (3 * Cc) + Cc + h * Dd;
    async_copy_b128(&Ks[buf][kr0 * 72 + dc0], base + (size_t)kr0 * (3 * Cc) + dc0);
    async_copy_b128(&Ks[buf][kr1 * 72 + dc1], base + (size_t)kr1 * (3 * Cc) + dc1);
  };
  auto load_V = [&](int k0, v8bf* vreg) {
    const __bf16* base = qkv + (size_t)(bb * Tt + k0) * (3 * Cc) + 2 * Cc + h * Dd;
    vreg[0] = *(const v8bf*)(base + (size_t)kr0 * (3 * Cc) + dc0);
    vreg[1] = *(const v8bf*)(base + (size_t)kr1 * (3 * Cc) + dc1);
  };
  auto store_V = [&](int buf, const v8bf* vreg) {
#pragma unroll
    for (int j = 0; j < 8; ++j) Vs[buf][(dc0 + j) * 40 + kr0] = vreg[0][j];
#pragma unroll
    for (int j = 0; j < 8; ++j) Vs[buf][(dc1 + j) * 40 + kr1] = vreg[1][j];
  };

  // Q fragments (two d-halves), loaded directly from global
  Frag qf[2];
  {
    const __bf16* qp = qkv + (size_t)(bb * Tt + qbase + lr) * (3 * Cc) + h * Dd;
#pragma unroll
    for (int s = 0; s < 2; ++s) {
      qf[s].h[0] = *(const v8bf*)(qp + s * 32 + kbA);
      qf[s].h[1] = *(const v8bf*)(qp + s * 32 + kbA + 16);
    }
  }

  const v8f vzero = {0.f, 0.f, 0.f, 0.f, 0.f, 0.f, 0.f, 0.f};
  v8f o[4];
#pragma unroll
  for (int nt = 0; nt < 4; ++nt) o[nt] = vzero;
  float mrow[8], lrow[8];
#pragma unroll
  for (int v = 0; v < 8; ++v) { mrow[v] = -1e30f; lrow[v] = 0.f; }

  const int kend = (qt + 1) * 64;   // block-uniform causal bound

  // ---- prologue: fill buffer 0 ----
  v8bf vreg[2];
  issue_async_K(0, 0);
  load_V(0, vreg);
  store_V(0, vreg);
  wait_async();
  __syncthreads();

  int buf = 0;
  for (int k0 = 0; k0 < kend; k0 += 32) {
    const int k1 = k0 + 32;
    const bool more = (k1 < kend);
    if (more) {
      issue_async_K(buf ^ 1, k1);
      load_V(k1, vreg);
    }

    if (k0 <= qbase + 15) {   // tile intersects causal region for this wave
      // ---- S = Q * K^T (two 16-col tiles, contraction over d=64) ----
      v8f st[2];
#pragma unroll
      for (int nt = 0; nt < 2; ++nt) {
        v8f s = vzero;
#pragma unroll
        for (int ds = 0; ds < 2; ++ds) {
          Frag bfr;
          const __bf16* p = Ks[buf] + (nt * 16 + lr) * 72 + ds * 32;
          bfr.h[0] = *(const v8bf*)(p + kbA);
          bfr.h[1] = *(const v8bf*)(p + kbA + 16);
          s = __builtin_amdgcn_wmma_f32_16x16x32_bf16(
              false, qf[ds].v, false, bfr.v, (short)0, s, false, false);
        }
        st[nt] = s;
      }
      // ---- online softmax (butterfly within 16-lane groups) ----
#pragma unroll
      for (int v = 0; v < 8; ++v) {
        int rowg = qbase + v + 8 * hi;
        float s0 = st[0][v] * scale;
        float s1 = st[1][v] * scale;
        if (k0 + lr > rowg)      s0 = -1e30f;
        if (k0 + 16 + lr > rowg) s1 = -1e30f;
        float mx = fmaxf(s0, s1);
#pragma unroll
        for (int off = 1; off < 16; off <<= 1) mx = fmaxf(mx, __shfl_xor(mx, off, 32));
        float mnew  = fmaxf(mrow[v], mx);
        float alpha = __expf(mrow[v] - mnew);
        mrow[v] = mnew;
        float p0 = __expf(s0 - mnew);
        float p1 = __expf(s1 - mnew);
        float sm = p0 + p1;
#pragma unroll
        for (int off = 1; off < 16; off <<= 1) sm += __shfl_xor(sm, off, 32);
        lrow[v] = lrow[v] * alpha + sm;
#pragma unroll
        for (int nt = 0; nt < 4; ++nt) o[nt][v] *= alpha;
        int prow = v + 8 * hi;
        Ps[wave][prow * 40 + lr]      = (__bf16)p0;
        Ps[wave][prow * 40 + 16 + lr] = (__bf16)p1;
      }
      // per-wave LDS RAW fence (CDNA5 split counter)
      asm volatile("s_wait_dscnt 0x0" ::: "memory");

      // ---- O += P * V ----
      Frag pf;
      {
        const __bf16* p = Ps[wave] + lr * 40;
        pf.h[0] = *(const v8bf*)(p + kbA);
        pf.h[1] = *(const v8bf*)(p + kbA + 16);
      }
#pragma unroll
      for (int nt = 0; nt < 4; ++nt) {
        Frag bfr;
        const __bf16* p = Vs[buf] + (nt * 16 + lr) * 40;
        bfr.h[0] = *(const v8bf*)(p + kbA);
        bfr.h[1] = *(const v8bf*)(p + kbA + 16);
        o[nt] = __builtin_amdgcn_wmma_f32_16x16x32_bf16(
            false, pf.v, false, bfr.v, (short)0, o[nt], false, false);
      }
    }

    if (more) {
      store_V(buf ^ 1, vreg);
      wait_async();
    }
    __syncthreads();
    buf ^= 1;
  }

  // normalize and write y (bf16, row-major [B,T,C] with head offset)
#pragma unroll
  for (int nt = 0; nt < 4; ++nt) {
#pragma unroll
    for (int v = 0; v < 8; ++v) {
      int rowg = qbase + v + 8 * hi;
      int d = nt * 16 + lr;
      y[(size_t)(bb * Tt + rowg) * Cc + h * Dd + d] = (__bf16)(o[nt][v] / lrow[v]);
    }
  }
}

// ---------------------------------------------------------------------------
// Host-side orchestration
// ---------------------------------------------------------------------------
extern "C" void kernel_launch(void* const* d_in, const int* in_sizes, int n_in,
                              void* d_out, int out_size, void* d_ws, size_t ws_size,
                              hipStream_t stream) {
  const float* x    = (const float*)d_in[0];
  const float* Wqkv = (const float*)d_in[1];
  const float* bqkv = (const float*)d_in[2];
  const float* Wo   = (const float*)d_in[3];
  const float* bo   = (const float*)d_in[4];
  const float* ln1w = (const float*)d_in[5];
  const float* ln1b = (const float*)d_in[6];
  const float* ln2w = (const float*)d_in[7];
  const float* ln2b = (const float*)d_in[8];
  const float* Wfc  = (const float*)d_in[9];
  const float* bfc  = (const float*)d_in[10];
  const float* Wp   = (const float*)d_in[11];
  const float* bp   = (const float*)d_in[12];

  char* ws = (char*)d_ws;
  size_t off = 0;
  auto alloc = [&](size_t bytes) {
    void* p = ws + off;
    off += (bytes + 255) & ~(size_t)255;
    return p;
  };
  __bf16* Wqkv_b = (__bf16*)alloc((size_t)Cc * 3 * Cc * 2);   //  6.0 MB
  __bf16* Wo_b   = (__bf16*)alloc((size_t)Cc * Cc * 2);       //  2.0 MB
  __bf16* Wfc_b  = (__bf16*)alloc((size_t)Cc * 4 * Cc * 2);   //  8.0 MB
  __bf16* Wp_b   = (__bf16*)alloc((size_t)4 * Cc * Cc * 2);   //  8.0 MB
  __bf16* h_b    = (__bf16*)alloc((size_t)BT * Cc * 2);       // 16 MB (h, reused as h2)
  __bf16* u_b    = (__bf16*)alloc((size_t)BT * 4 * Cc * 2);   // 64 MB (qkv, reused as m)
  __bf16* y_b    = (__bf16*)alloc((size_t)BT * Cc * 2);       // 16 MB
  float*  x2     = (float*) alloc((size_t)BT * Cc * 4);       // 32 MB
  __bf16* qkv_b  = u_b;
  __bf16* m_b    = u_b;

  // 1. convert weights fp32 -> bf16
  {
    int n;
    n = Cc * 3 * Cc; cvt_f32_bf16<<<(n + 255) / 256, 256, 0, stream>>>(Wqkv, Wqkv_b, n);
    n = Cc * Cc;     cvt_f32_bf16<<<(n + 255) / 256, 256, 0, stream>>>(Wo,   Wo_b,   n);
    n = Cc * 4 * Cc; cvt_f32_bf16<<<(n + 255) / 256, 256, 0, stream>>>(Wfc,  Wfc_b,  n);
    n = 4 * Cc * Cc; cvt_f32_bf16<<<(n + 255) / 256, 256, 0, stream>>>(Wp,   Wp_b,   n);
  }

  // 2. h = LN1(x)
  ln_kernel<<<BT, 256, 0, stream>>>(x, ln1w, ln1b, h_b);

  // 3. qkv = h @ Wqkv + bqkv   (bf16 out)
  gemm_bf16_kernel<true, false, false, true>
      <<<dim3(3 * Cc / 128, BT / 128), 256, 0, stream>>>(
      h_b, Wqkv_b, bqkv, nullptr, nullptr, qkv_b, BT, 3 * Cc, Cc);

  // 4. y = attention(qkv)
  attn_kernel<<<Bb * Hh * (Tt / 64), 128, 0, stream>>>(qkv_b, y_b);

  // 5. x2 = x + y @ Wo + bo   (fp32 out, fp32 residual)
  gemm_bf16_kernel<true, true, false, false>
      <<<dim3(Cc / 128, BT / 128), 256, 0, stream>>>(
      y_b, Wo_b, bo, x, x2, nullptr, BT, Cc, Cc);

  // 6. h2 = LN2(x2)   (reuse h buffer)
  ln_kernel<<<BT, 256, 0, stream>>>(x2, ln2w, ln2b, h_b);

  // 7. m = gelu(h2 @ Wfc + bfc)   (bf16 out, reuse qkv buffer)
  gemm_bf16_kernel<true, false, true, true>
      <<<dim3(4 * Cc / 128, BT / 128), 256, 0, stream>>>(
      h_b, Wfc_b, bfc, nullptr, nullptr, m_b, BT, 4 * Cc, Cc);

  // 8. out = x2 + m @ Wp + bp   (fp32 out = d_out)
  gemm_bf16_kernel<true, true, false, false>
      <<<dim3(Cc / 128, BT / 128), 256, 0, stream>>>(
      m_b, Wp_b, bp, x2, (float*)d_out, nullptr, BT, Cc, 4 * Cc);
}
